// MoEMLP_55061480735482
// MI455X (gfx1250) — compile-verified, matrix operand order checked by reading, not run on hardware
//
#include <hip/hip_runtime.h>

// Problem constants (from reference): B=8,S=2048,H=512,F=1024,E=16,K=2
#define BB 8
#define SS 2048
#define HH 512
#define FF 1024
#define EE 16
#define KK 2
#define NN (BB * SS)     // 16384 tokens
#define NH (NN * HH)     // 8388608 output elements (then +1 for l_aux)

// Packed-weight geometry
#define KT1 (HH / 32)    // 16 k-tiles for GEMM1
#define NT1 (FF / 16)    // 64 n-tiles for GEMM1
#define KT2 (FF / 32)    // 32 k-tiles for GEMM2
#define NT2 (HH / 16)    // 32 n-tiles for GEMM2
#define EXP_STRIDE (FF * HH)   // 524288 bf16 elements per expert (both W1p and W2p)

typedef __attribute__((ext_vector_type(16))) __bf16 v16bf;
typedef __attribute__((ext_vector_type(8)))  float  v8f;

union Frag {
  v16bf v;
  uint4          q[2];
  unsigned int   w[8];
  unsigned short u[16];
};

__device__ __forceinline__ unsigned short f2bf(float f) {
  union { float f; unsigned int u; } a; a.f = f;
  unsigned int r = a.u + 0x7FFFu + ((a.u >> 16) & 1u);  // round-to-nearest-even
  return (unsigned short)(r >> 16);
}

// ISA 16-bit A/B fragment layout (16x32 / 32x16):
// lanes 0-15: VGPR v<4 -> K = 2v,2v+1 ; v>=4 -> K = 16+2(v-4) ; lanes 16-31: +8
__device__ __forceinline__ int kbase(int v, int hs) {
  return ((v < 4) ? (v * 2) : (16 + (v - 4) * 2)) + hs * 8;
}

__device__ __forceinline__ float gelu_tanh(float x) {
  const float c = 0.7978845608028654f;
  float t = tanhf(c * (x + 0.044715f * x * x * x));
  return 0.5f * x * (1.0f + t);
}

// --- fragment load/MMA helpers ---------------------------------------------
template <int CNT>
__device__ __forceinline__ void load_bfrags(Frag (&b)[CNT], const unsigned short* we,
                                            int kt, int NT, int ntile0, int lane) {
  #pragma unroll
  for (int nt = 0; nt < CNT; ++nt) {
    const uint4* bp = (const uint4*)(we + ((size_t)(kt * NT + ntile0 + nt) * 32 + lane) * 16);
    b[nt].q[0] = bp[0];
    b[nt].q[1] = bp[1];
  }
}

__device__ __forceinline__ void load_afrag(Frag& a, const unsigned short* pk,
                                           int kt, int lane) {
  const uint4* ap = (const uint4*)(pk + ((size_t)kt * 32 + lane) * 16);
  a.q[0] = ap[0];
  a.q[1] = ap[1];
}

template <int CNT>
__device__ __forceinline__ void mma_all(v8f (&acc)[CNT], const Frag& a, const Frag (&b)[CNT]) {
  #pragma unroll
  for (int nt = 0; nt < CNT; ++nt)
    acc[nt] = __builtin_amdgcn_wmma_f32_16x16x32_bf16(
        false, a.v, false, b[nt].v, (short)0, acc[nt], false, false);
}

// ---------------------------------------------------------------------------
__global__ void zero_kernel(float* p, unsigned long long n) {
  unsigned long long i  = (unsigned long long)blockIdx.x * blockDim.x + threadIdx.x;
  unsigned long long st = (unsigned long long)gridDim.x * blockDim.x;
  for (; i < n; i += st) p[i] = 0.f;
}

// ---------------------------------------------------------------------------
// Pack W1 [E,F,H] fp32 -> bf16 B-fragments. One thread per bf16 *pair*.
__global__ __launch_bounds__(256) void pack_w1_kernel(const float* __restrict__ W1,
                                                      unsigned short* __restrict__ W1p) {
  unsigned int i = blockIdx.x * 256 + threadIdx.x;   // 2^22 pairs total
  int v    = i & 7;
  int lane = (i >> 3) & 31;
  int nt   = (i >> 8) & 63;
  int kt   = (i >> 14) & 15;
  int e    = i >> 18;
  int n = nt * 16 + (lane & 15);
  int k = kt * 32 + kbase(v, lane >> 4);
  float2 f2 = *(const float2*)(W1 + ((size_t)(e * FF + n)) * HH + k);
  unsigned int pr = (unsigned int)f2bf(f2.x) | ((unsigned int)f2bf(f2.y) << 16);
  ((unsigned int*)W1p)[i] = pr;
}

// Pack W2 [E,H,F] fp32 -> bf16 B-fragments.
__global__ __launch_bounds__(256) void pack_w2_kernel(const float* __restrict__ W2,
                                                      unsigned short* __restrict__ W2p) {
  unsigned int i = blockIdx.x * 256 + threadIdx.x;   // 2^22 pairs total
  int v    = i & 7;
  int lane = (i >> 3) & 31;
  int nt   = (i >> 8) & 31;
  int kt   = (i >> 13) & 31;
  int e    = i >> 18;
  int n = nt * 16 + (lane & 15);
  int k = kt * 32 + kbase(v, lane >> 4);
  float2 f2 = *(const float2*)(W2 + ((size_t)(e * HH + n)) * FF + k);
  unsigned int pr = (unsigned int)f2bf(f2.x) | ((unsigned int)f2bf(f2.y) << 16);
  ((unsigned int*)W2p)[i] = pr;
}

// ---------------------------------------------------------------------------
// Router: 16 tokens per 256-thread block; thread (t,e) = one dot product.
__global__ __launch_bounds__(256) void router_kernel(
    const float* __restrict__ x, const float* __restrict__ Wr,
    int* __restrict__ cnt, float* __restrict__ Psum,
    int* __restrict__ list, float* __restrict__ wgt)
{
  __shared__ float lg[16][EE + 1];
  __shared__ float Pacc[EE];
  const int tid   = threadIdx.x;
  const int t     = tid >> 4;
  const int e     = tid & 15;
  const int token = blockIdx.x * 16 + t;

  const float* xr = x + (size_t)token * HH;
  const float* wr = Wr + e * HH;
  float s = 0.f;
  #pragma unroll 4
  for (int i = 0; i < HH; i += 4) {
    float4 a = *(const float4*)(xr + i);
    float4 b = *(const float4*)(wr + i);
    s += a.x * b.x + a.y * b.y + a.z * b.z + a.w * b.w;
  }
  lg[t][e] = s;
  if (tid < EE) Pacc[tid] = 0.f;
  __syncthreads();

  if (e == 0) {  // 16 threads, one per token
    float v0 = -3.4e38f, v1 = -3.4e38f; int i0 = 0, i1 = 0;
    for (int j = 0; j < EE; ++j) {
      float l = lg[t][j];
      if (l > v0)      { v1 = v0; i1 = i0; v0 = l; i0 = j; }
      else if (l > v1) { v1 = l; i1 = j; }
    }
    float Z = 0.f, pbuf[EE];
    for (int j = 0; j < EE; ++j) { float p = __expf(lg[t][j] - v0); pbuf[j] = p; Z += p; }
    float invZ = 1.f / Z;
    for (int j = 0; j < EE; ++j) atomicAdd(&Pacc[j], pbuf[j] * invZ);
    float p0 = 1.f / (1.f + __expf(v1 - v0));
    float p1 = 1.f - p0;
    int q0 = atomicAdd(&cnt[i0], 1);
    list[i0 * NN + q0] = token; wgt[i0 * NN + q0] = p0;
    int q1 = atomicAdd(&cnt[i1], 1);
    list[i1 * NN + q1] = token; wgt[i1 * NN + q1] = p1;
  }
  __syncthreads();
  if (tid < EE) atomicAdd(&Psum[tid], Pacc[tid]);
}

// ---------------------------------------------------------------------------
// Grouped expert GEMM: block = (expert e, 16-token tile). 8 waves.
// B fragments double-buffered over k-tiles so a full 8-WMMA batch overlaps
// the next batch's 16 global_load_b128s.
__global__ __launch_bounds__(256) void moe_expert_kernel(
    const float* __restrict__ x,
    const unsigned short* __restrict__ W1p, const float* __restrict__ b1,
    const unsigned short* __restrict__ W2p, const float* __restrict__ b2,
    const int* __restrict__ cnt, const int* __restrict__ list, const float* __restrict__ wgt,
    float* __restrict__ y)
{
  const int e     = blockIdx.x;
  const int tile  = blockIdx.y;
  const int count = cnt[e];
  if (tile * 16 >= count) return;   // uniform early-exit

  __shared__ unsigned short xpack[KT1][32][16];  // x tile, A-fragment order (16 KB)
  __shared__ unsigned short hpack[KT2][32][16];  // h tile, A-fragment order (32 KB)
  __shared__ int   tok_s[16];
  __shared__ float wt_s[16];

  const int tid  = threadIdx.x;
  const int wave = tid >> 5;
  const int lane = tid & 31;
  const int m    = lane & 15;   // A row / B column / C-D column index
  const int hsel = lane >> 4;   // lane-half selector

  if (tid < 16) {
    int p = tile * 16 + tid;
    if (p < count) { tok_s[tid] = list[e * NN + p]; wt_s[tid] = wgt[e * NN + p]; }
    else           { tok_s[tid] = -1;               wt_s[tid] = 0.f; }
  }
  __syncthreads();

  // Fill xpack directly in fragment order: one bf16 pair per iteration.
  for (int i = tid; i < KT1 * 32 * 8; i += 256) {
    int v  = i & 7;
    int ln = (i >> 3) & 31;
    int kt = i >> 8;
    int row = ln & 15;
    int k   = kt * 32 + kbase(v, ln >> 4);
    int t   = tok_s[row];
    unsigned int pr = 0;
    if (t >= 0) {
      float2 f2 = *(const float2*)(x + (size_t)t * HH + k);
      pr = (unsigned int)f2bf(f2.x) | ((unsigned int)f2bf(f2.y) << 16);
    }
    *(unsigned int*)&xpack[kt][ln][2 * v] = pr;
  }
  __syncthreads();

  // ---------------- GEMM1: h = gelu(x @ W1[e]^T + b1), wave owns 128 F-cols ----
  v8f acc1[8] = {};
  {
    const unsigned short* w1e = W1p + (size_t)e * EXP_STRIDE;
    const unsigned short* xp  = &xpack[0][0][0];
    const int nt0 = wave * 8;
    Frag b0[8], b1f[8], a;
    load_bfrags(b0, w1e, 0, NT1, nt0, lane);
    for (int kt = 0; kt < KT1; kt += 2) {
      load_bfrags(b1f, w1e, kt + 1, NT1, nt0, lane);   // prefetch next k-tile
      load_afrag(a, xp, kt, lane);
      mma_all(acc1, a, b0);
      if (kt + 2 < KT1) load_bfrags(b0, w1e, kt + 2, NT1, nt0, lane);
      load_afrag(a, xp, kt + 1, lane);
      mma_all(acc1, a, b1f);
    }
  }

  // Bias + GELU, store h into LDS already in GEMM2 A-fragment order.
  #pragma unroll
  for (int nt = 0; nt < 8; ++nt) {
    const int fcol = wave * 128 + nt * 16 + m;
    const float bias = b1[e * FF + fcol];
    const int kt2 = fcol >> 5;
    const int ko  = fcol & 31;
    const int p2  = ko & 1;
    const int qq  = ko >> 1;
    const int v2  = (qq < 8) ? (qq & 3) : (4 + ((qq - 8) & 3));
    const int hs2 = (qq < 8) ? (qq >> 2) : ((qq - 8) >> 2);
    #pragma unroll
    for (int d = 0; d < 8; ++d) {
      int row = d + 8 * hsel;               // C/D layout: M = d + 8*(lane/16)
      hpack[kt2][row + 16 * hs2][2 * v2 + p2] = f2bf(gelu_tanh(acc1[nt][d] + bias));
    }
  }
  __syncthreads();

  // ---------------- GEMM2: o = h @ W2[e]^T + b2, wave owns 64 H-cols ----------
  v8f acc2[4] = {};
  {
    const unsigned short* w2e = W2p + (size_t)e * EXP_STRIDE;
    const unsigned short* hp  = &hpack[0][0][0];
    const int nt0 = wave * 4;
    Frag b0[4], b1f[4], a;
    load_bfrags(b0, w2e, 0, NT2, nt0, lane);
    for (int kt = 0; kt < KT2; kt += 2) {
      load_bfrags(b1f, w2e, kt + 1, NT2, nt0, lane);   // prefetch next k-tile
      load_afrag(a, hp, kt, lane);
      mma_all(acc2, a, b0);
      if (kt + 2 < KT2) load_bfrags(b0, w2e, kt + 2, NT2, nt0, lane);
      load_afrag(a, hp, kt + 1, lane);
      mma_all(acc2, a, b1f);
    }
  }

  // Bias + weighted atomic scatter into y.
  #pragma unroll
  for (int nt = 0; nt < 4; ++nt) {
    const int hcol = wave * 64 + nt * 16 + m;
    const float bias = b2[e * HH + hcol];
    #pragma unroll
    for (int d = 0; d < 8; ++d) {
      int row = d + 8 * hsel;
      int t = tok_s[row];
      if (t >= 0)
        atomicAdd(&y[(size_t)t * HH + hcol], wt_s[row] * (acc2[nt][d] + bias));
    }
  }
}

// ---------------------------------------------------------------------------
__global__ void laux_kernel(const int* __restrict__ cnt, const float* __restrict__ Psum,
                            float* __restrict__ out) {
  int e = threadIdx.x;
  float v = 0.f;
  if (e < EE) {
    float f = (float)cnt[e] / (float)(NN * KK);
    float P = Psum[e] / (float)NN;
    v = f * P;
  }
  for (int o = 8; o > 0; o >>= 1) v += __shfl_down(v, o, 32);
  if (e == 0) out[0] = (float)EE * v;
}

// ---------------------------------------------------------------------------
extern "C" void kernel_launch(void* const* d_in, const int* in_sizes, int n_in,
                              void* d_out, int out_size, void* d_ws, size_t ws_size,
                              hipStream_t stream) {
  const float* x  = (const float*)d_in[0];
  const float* Wr = (const float*)d_in[1];
  const float* W1 = (const float*)d_in[2];
  const float* b1 = (const float*)d_in[3];
  const float* W2 = (const float*)d_in[4];
  const float* b2 = (const float*)d_in[5];
  (void)in_sizes; (void)n_in; (void)ws_size;   // k hardcoded = 2 (reference constant)

  float* y = (float*)d_out;

  // Workspace layout (16B-aligned blocks):
  //   W1p: E*F*H bf16 (16.78 MB) | W2p: E*H*F bf16 (16.78 MB)
  //   cnt[E] | Psum[E] | list[E*N] | wgt[E*N]   (~2.1 MB)
  unsigned short* W1p = (unsigned short*)d_ws;
  unsigned short* W2p = W1p + (size_t)EE * EXP_STRIDE;
  int*   cnt  = (int*)(W2p + (size_t)EE * EXP_STRIDE);
  float* Psum = (float*)(cnt + EE);
  int*   list = (int*)(Psum + EE);
  float* wgt  = (float*)(list + (size_t)EE * NN);

  const unsigned int pack_pairs = (unsigned int)EE * EXP_STRIDE / 2;  // 2^22

  zero_kernel<<<2048, 256, 0, stream>>>(y, (unsigned long long)out_size);
  zero_kernel<<<1, 32, 0, stream>>>((float*)cnt, 2ull * EE);
  pack_w1_kernel<<<pack_pairs / 256, 256, 0, stream>>>(W1, W1p);
  pack_w2_kernel<<<pack_pairs / 256, 256, 0, stream>>>(W2, W2p);
  router_kernel<<<NN / 16, 256, 0, stream>>>(x, Wr, cnt, Psum, list, wgt);
  moe_expert_kernel<<<dim3(EE, NN / 16), 256, 0, stream>>>(x, W1p, b1, W2p, b2,
                                                           cnt, list, wgt, y);
  laux_kernel<<<1, 32, 0, stream>>>(cnt, Psum, y + NH);
}
